// RGCN_55405078118544
// MI455X (gfx1250) — compile-verified, matrix operand order checked by reading, not run on hardware
//
#include <hip/hip_runtime.h>
#include <hip/hip_bf16.h>

// ---------------------------------------------------------------------------
// RGCN (basis-decomposed, mean aggregation) for MI455X / gfx1250
//   - all GEMMs via v_wmma_f32_16x16x32_bf16, wave32
//   - operands pre-packed into WMMA fragment order (bf16); A packed once,
//     reused by 5 GEMMs; B staged per-workgroup in LDS (shared by 8 waves)
//   - edge phase: per-(rel,dst) degree counted once (shared by both layers),
//     1/deg folded into the scatter so relation-sum is one atomic stream
// ---------------------------------------------------------------------------

#define NUM_MIRNA 20000
#define NUM_GENE  30000
#define NNODES    (NUM_MIRNA + NUM_GENE)   // 50000
#define NREL      4
#define NEDGE     1000000
#define IN_C      128
#define HID_C     128
#define OUT_C     64
#define NBASIS    10

typedef __attribute__((ext_vector_type(16))) __bf16 v16bf;
typedef __attribute__((ext_vector_type(8)))  float  v8f;

// ---------------------------------------------------------------------------
// Fragment packing.  WMMA bf16 A-fragment (16x32 tile), lane l:
//   half = l>>4, r = l&15 ; element i<8  -> k = ks*32 + half*8 + i
//                           element i>=8 -> k = ks*32 + 16 + half*8 + (i-8)
// A: row = strip*16 + r ; B: col = tile*16 + r (B mirrors A with M->N).
// Packed layout: frag (unit = strip*KSTEPS+ks  or  tile*KSTEPS+ks) stored as
// 32 lanes x 16 bf16 contiguous -> GEMM reads one aligned 32B vector per frag.
// ---------------------------------------------------------------------------
__global__ __launch_bounds__(256)
void pack_A(const float* __restrict__ A, __bf16* __restrict__ Ap,
            int Mstrips, int K)
{
    const int gid    = blockIdx.x * blockDim.x + threadIdx.x;
    const int ksteps = K >> 5;
    const int unit   = gid >> 5, lane = gid & 31;
    if (unit >= Mstrips * ksteps) return;
    const int strip = unit / ksteps, ks = unit - strip * ksteps;
    const int half = lane >> 4, idx16 = lane & 15;
    const float* __restrict__ ar =
        A + (size_t)(strip * 16 + idx16) * K + ks * 32 + half * 8;
    v16bf v;
    #pragma unroll
    for (int i = 0; i < 8; ++i) {
        v[i]     = (__bf16)ar[i];
        v[i + 8] = (__bf16)ar[16 + i];
    }
    *(v16bf*)(Ap + ((size_t)unit * 32 + lane) * 16) = v;
}

__global__ __launch_bounds__(256)
void pack_B(const float* __restrict__ B, __bf16* __restrict__ Bp,
            int K, int N)
{
    const int gid    = blockIdx.x * blockDim.x + threadIdx.x;
    const int ksteps = K >> 5;
    const int unit   = gid >> 5, lane = gid & 31;
    if (unit >= (N >> 4) * ksteps) return;
    const int t = unit / ksteps, ks = unit - t * ksteps;
    const int half = lane >> 4, idx16 = lane & 15;
    const int col = t * 16 + idx16;
    const int kb  = ks * 32 + half * 8;
    v16bf v;
    #pragma unroll
    for (int i = 0; i < 8; ++i) {
        v[i]     = (__bf16)B[(size_t)(kb + i)      * N + col];
        v[i + 8] = (__bf16)B[(size_t)(kb + 16 + i) * N + col];
    }
    *(v16bf*)(Bp + ((size_t)unit * 32 + lane) * 16) = v;
}

// ---------------------------------------------------------------------------
// D[16-strip x NT*16] = Ap-strip @ Bp (+Cin) (+bias).  One wave per 16-row
// strip; B fragments staged once per workgroup in LDS and shared by all 8
// waves; A fragment loaded once per K-step, reused across NT column tiles.
// Inner loop: 1 A-vec global load + NT LDS vec loads + NT v_wmma.
// ---------------------------------------------------------------------------
template <int KSTEPS, int NT, bool HAS_CIN, bool HAS_BIAS>
__global__ __launch_bounds__(256)
void wmma_gemm_packed(const __bf16* __restrict__ Ap, const __bf16* __restrict__ Bp,
                      const float* __restrict__ Cin, const float* __restrict__ bias,
                      float* __restrict__ D, int Mstrips)
{
    constexpr int NCOLS  = NT * 16;
    constexpr int BELEMS = NT * KSTEPS * 32 * 16;          // bf16 elements

    __shared__ __attribute__((aligned(32))) __bf16 Bs[BELEMS];

    // cooperative B stage: global -> LDS, 16B chunks, all 256 threads
    {
        const float4* __restrict__ gsrc = (const float4*)Bp;
        float4* __restrict__ ldst = (float4*)Bs;
        constexpr int NV = BELEMS * 2 / 16;                // 16B vectors
        #pragma unroll 4
        for (int i = threadIdx.x; i < NV; i += 256)
            ldst[i] = gsrc[i];
    }
    __syncthreads();

    const int lane = threadIdx.x & 31;
    const int wave = threadIdx.x >> 5;
    const int tm   = blockIdx.x * (blockDim.x >> 5) + wave;
    const bool valid = tm < Mstrips;
    const int half = lane >> 4, idx16 = lane & 15;

    v8f acc[NT];
    #pragma unroll
    for (int t = 0; t < NT; ++t) acc[t] = (v8f){};

    if (valid) {
        const v16bf* __restrict__ ap =
            (const v16bf*)Ap + (size_t)tm * KSTEPS * 32 + lane;
        const v16bf* __restrict__ bs = (const v16bf*)Bs + lane;

        for (int ks = 0; ks < KSTEPS; ++ks) {
            const v16bf a = ap[ks * 32];
            #pragma unroll
            for (int t = 0; t < NT; ++t) {
                const v16bf b = bs[(t * KSTEPS + ks) * 32];
                acc[t] = __builtin_amdgcn_wmma_f32_16x16x32_bf16(
                    false, a, false, b, (short)0, acc[t], false, false);
            }
        }

        #pragma unroll
        for (int t = 0; t < NT; ++t) {
            const int col = t * 16 + idx16;
            const float bv = HAS_BIAS ? bias[col] : 0.0f;
            #pragma unroll
            for (int v = 0; v < 8; ++v) {
                const int row = tm * 16 + v + half * 8;
                const size_t o = (size_t)row * NCOLS + col;
                float val = acc[t][v] + bv;
                if (HAS_CIN) val += Cin[o];
                D[o] = val;
            }
        }
    }
}

// W[r] = sum_b comp[r,b] * basis[b]  (tiny)
__global__ void combine_basis(const float* __restrict__ comp,
                              const float* __restrict__ basis,
                              float* __restrict__ W, int IO)
{
    const int idx = blockIdx.x * blockDim.x + threadIdx.x;
    if (idx >= NREL * IO) return;
    const int r = idx / IO, io = idx - r * IO;
    float s = 0.0f;
    #pragma unroll
    for (int b = 0; b < NBASIS; ++b)
        s += comp[r * NBASIS + b] * basis[(size_t)b * IO + io];
    W[idx] = s;
}

// per-(relation,dst) in-degree
__global__ void edge_count(const int* __restrict__ etype,
                           const int* __restrict__ dst,
                           unsigned* __restrict__ cnt)
{
    const int e = blockIdx.x * blockDim.x + threadIdx.x;
    if (e >= NEDGE) return;
    atomicAdd(&cnt[(size_t)etype[e] * NNODES + dst[e]], 1u);
}

__global__ void make_recip(const unsigned* __restrict__ cnt,
                           float* __restrict__ recip)
{
    const int i = blockIdx.x * blockDim.x + threadIdx.x;
    if (i >= NREL * NNODES) return;
    const unsigned c = cnt[i];
    recip[i] = 1.0f / (float)(c > 0u ? c : 1u);
}

// agg[dst] += h[etype][src] * (1/deg(etype,dst)); relation-sum is implicit.
template <int C>
__global__ __launch_bounds__(256)
void scatter_mean(const float* __restrict__ h,
                  const int* __restrict__ src, const int* __restrict__ dst,
                  const int* __restrict__ etype,
                  const float* __restrict__ recip,
                  float* __restrict__ agg)
{
    constexpr int TPE = C / 4;
    constexpr int EPB = 256 / TPE;
    const int eLocal = threadIdx.x / TPE;
    const int lc     = threadIdx.x - eLocal * TPE;
    const long e = (long)blockIdx.x * EPB + eLocal;
    if (e >= NEDGE) return;
    const int s = src[e], d = dst[e], r = etype[e];
    const float w = recip[(size_t)r * NNODES + d];
    const float4 hv = *(const float4*)(h + ((size_t)r * NNODES + s) * C + lc * 4);
    float* __restrict__ p = agg + (size_t)d * C + lc * 4;
    atomicAdd(p + 0, hv.x * w);
    atomicAdd(p + 1, hv.y * w);
    atomicAdd(p + 2, hv.z * w);
    atomicAdd(p + 3, hv.w * w);
}

// ---------------------------------------------------------------------------
static inline void launch_pack_A(const float* A, __bf16* Ap, int M, int K,
                                 hipStream_t s)
{
    const int units = (M / 16) * (K / 32);
    pack_A<<<(units * 32 + 255) / 256, 256, 0, s>>>(A, Ap, M / 16, K);
}
static inline void launch_pack_B(const float* B, __bf16* Bp, int K, int N,
                                 hipStream_t s)
{
    const int units = (N / 16) * (K / 32);
    pack_B<<<(units * 32 + 255) / 256, 256, 0, s>>>(B, Bp, K, N);
}
template <int KSTEPS, int NT, bool HAS_CIN, bool HAS_BIAS>
static inline void launch_gemm(const __bf16* Ap, const __bf16* Bp,
                               const float* Cin, const float* bias, float* D,
                               int M, hipStream_t s)
{
    const int strips = M / 16;
    wmma_gemm_packed<KSTEPS, NT, HAS_CIN, HAS_BIAS>
        <<<(strips + 7) / 8, 256, 0, s>>>(Ap, Bp, Cin, bias, D, strips);
}

extern "C" void kernel_launch(void* const* d_in, const int* in_sizes, int n_in,
                              void* d_out, int out_size, void* d_ws, size_t ws_size,
                              hipStream_t stream)
{
    const float* x_mirna = (const float*)d_in[0];
    const float* x_gene  = (const float*)d_in[1];
    const int*   eidx    = (const int*)d_in[2];
    const int*   etype   = (const int*)d_in[3];
    const float* lin_w_m = (const float*)d_in[4];
    const float* lin_b_m = (const float*)d_in[5];
    const float* lin_w_g = (const float*)d_in[6];
    const float* lin_b_g = (const float*)d_in[7];
    const float* comp1   = (const float*)d_in[8];
    const float* basis1  = (const float*)d_in[9];
    const float* root1   = (const float*)d_in[10];
    const float* bias1   = (const float*)d_in[11];
    const float* comp2   = (const float*)d_in[12];
    const float* basis2  = (const float*)d_in[13];
    const float* root2   = (const float*)d_in[14];
    const float* bias2   = (const float*)d_in[15];

    const int* src = eidx;          // edge_index[0]
    const int* dst = eidx + NEDGE;  // edge_index[1]

    // ---- workspace carving -------------------------------------------------
    char* ws = (char*)d_ws;
    float*    x0    = (float*)ws;  ws += (size_t)NNODES * IN_C  * 4;          // 25.6 MB
    float*    x1    = (float*)ws;  ws += (size_t)NNODES * HID_C * 4;          // 25.6 MB
    float*    h     = (float*)ws;  ws += (size_t)NREL * NNODES * HID_C * 4;   // 102.4 MB
    float*    agg   = (float*)ws;  ws += (size_t)NNODES * HID_C * 4;          // 25.6 MB
    __bf16*   x0p   = (__bf16*)ws; ws += (size_t)NNODES * IN_C  * 2;          // 12.8 MB
    __bf16*   x1p   = (__bf16*)ws; ws += (size_t)NNODES * HID_C * 2;          // 12.8 MB
    float*    W1    = (float*)ws;  ws += (size_t)NREL * IN_C  * HID_C * 4;
    float*    W2    = (float*)ws;  ws += (size_t)NREL * HID_C * OUT_C * 4;
    __bf16*   wmP   = (__bf16*)ws; ws += (size_t)256 * IN_C * 2;              // lin_w_m packed
    __bf16*   wgP   = (__bf16*)ws; ws += (size_t)512 * IN_C * 2;              // lin_w_g packed
    __bf16*   W1p   = (__bf16*)ws; ws += (size_t)NREL * IN_C  * HID_C * 2;
    __bf16*   r1P   = (__bf16*)ws; ws += (size_t)IN_C * HID_C * 2;
    __bf16*   W2p   = (__bf16*)ws; ws += (size_t)NREL * HID_C * OUT_C * 2;
    __bf16*   r2P   = (__bf16*)ws; ws += (size_t)HID_C * OUT_C * 2;
    unsigned* cnt   = (unsigned*)ws; ws += (size_t)NREL * NNODES * 4;
    float*    recip = (float*)ws;    ws += (size_t)NREL * NNODES * 4;
    // packed miRNA/gene inputs alias the not-yet-live h region
    __bf16*   xmP   = (__bf16*)h;                               // 10.24 MB
    __bf16*   xgP   = (__bf16*)h + (size_t)NUM_MIRNA * 256;     // 30.72 MB

    // ---- relation weights from basis decomposition (tiny) ------------------
    combine_basis<<<(NREL * IN_C * HID_C + 255) / 256, 256, 0, stream>>>(comp1, basis1, W1, IN_C * HID_C);
    combine_basis<<<(NREL * HID_C * OUT_C + 255) / 256, 256, 0, stream>>>(comp2, basis2, W2, HID_C * OUT_C);

    // ---- pack all weight matrices to bf16 fragment order -------------------
    launch_pack_B(lin_w_m, wmP, 256, IN_C, stream);
    launch_pack_B(lin_w_g, wgP, 512, IN_C, stream);
    for (int r = 0; r < NREL; ++r) {
        launch_pack_B(W1 + (size_t)r * IN_C * HID_C, W1p + (size_t)r * IN_C * HID_C, IN_C, HID_C, stream);
        launch_pack_B(W2 + (size_t)r * HID_C * OUT_C, W2p + (size_t)r * HID_C * OUT_C, HID_C, OUT_C, stream);
    }
    launch_pack_B(root1, r1P, IN_C, HID_C, stream);
    launch_pack_B(root2, r2P, HID_C, OUT_C, stream);

    // ---- per-(relation,dst) mean normalizers: shared by both layers --------
    hipMemsetAsync(cnt, 0, (size_t)NREL * NNODES * 4, stream);
    edge_count<<<(NEDGE + 255) / 256, 256, 0, stream>>>(etype, dst, cnt);
    make_recip<<<(NREL * NNODES + 255) / 256, 256, 0, stream>>>(cnt, recip);

    // ---- input projections -> x0 [N,128] -----------------------------------
    launch_pack_A(x_mirna, xmP, NUM_MIRNA, 256, stream);
    launch_pack_A(x_gene,  xgP, NUM_GENE,  512, stream);
    launch_gemm<8,  8, false, true>(xmP, wmP, nullptr, lin_b_m, x0, NUM_MIRNA, stream);
    launch_gemm<16, 8, false, true>(xgP, wgP, nullptr, lin_b_g,
                                    x0 + (size_t)NUM_MIRNA * IN_C, NUM_GENE, stream);

    // ---- layer 1: h_r = x0 @ W1_r ; scatter-mean ; + x0 @ root1 + bias1 ----
    launch_pack_A(x0, x0p, NNODES, IN_C, stream);     // packed once, used 5x
    for (int r = 0; r < NREL; ++r)
        launch_gemm<4, 8, false, false>(x0p, W1p + (size_t)r * IN_C * HID_C, nullptr, nullptr,
                                        h + (size_t)r * NNODES * HID_C, NNODES, stream);
    hipMemsetAsync(agg, 0, (size_t)NNODES * HID_C * 4, stream);
    scatter_mean<HID_C><<<(NEDGE + 7) / 8, 256, 0, stream>>>(h, src, dst, etype, recip, agg);
    launch_gemm<4, 8, true, true>(x0p, r1P, agg, bias1, x1, NNODES, stream);

    // ---- layer 2: 128 -> 64, writes d_out directly -------------------------
    launch_pack_A(x1, x1p, NNODES, HID_C, stream);    // packed once, used 5x
    for (int r = 0; r < NREL; ++r)
        launch_gemm<4, 4, false, false>(x1p, W2p + (size_t)r * HID_C * OUT_C, nullptr, nullptr,
                                        h + (size_t)r * NNODES * OUT_C, NNODES, stream);
    hipMemsetAsync(agg, 0, (size_t)NNODES * OUT_C * 4, stream);
    scatter_mean<OUT_C><<<(NEDGE + 15) / 16, 256, 0, stream>>>(h, src, dst, etype, recip, agg);
    launch_gemm<4, 4, true, true>(x1p, r2P, agg, bias2, (float*)d_out, NNODES, stream);
}
